// CSRNet_5454608466634
// MI455X (gfx1250) — matrix operand order repaired and checked by reading, not compiled, not run on hardware
//
#include <hip/hip_runtime.h>

typedef __attribute__((ext_vector_type(16))) _Float16 v16h;
typedef __attribute__((ext_vector_type(8)))  float    v8f;

__device__ __forceinline__ v8f wmma_f16(v16h a, v16h b, v8f c) {
  return __builtin_amdgcn_wmma_f32_16x16x32_f16(false, a, false, b, (short)0, c, false, false);
}

// A fragment (16x32 f16): lane holds row m=lane&15; element e -> k = half*16 + e.
__device__ __forceinline__ v16h load_a_frag(const _Float16* As, int ldk) {
  int lane = threadIdx.x & 31;
  int half = lane >> 4;
  const _Float16* row = As + (size_t)(lane & 15) * ldk;
  v16h a;
#pragma unroll
  for (int ep = 0; ep < 8; ++ep) {
    int e = ep * 2;
    int k = half * 16 + e;
    a[e]     = row[k];
    a[e + 1] = row[k + 1];
  }
  return a;
}

// B fragment (32x16 f16) staged as [n][k]: lane n=lane&15, k = half*16 + e.
__device__ __forceinline__ v16h load_b_frag(const _Float16* Bs, int ldk) {
  int lane = threadIdx.x & 31;
  const _Float16* row = Bs + (size_t)(lane & 15) * ldk + (lane >> 4) * 16;
  v16h b;
#pragma unroll
  for (int e = 0; e < 16; ++e) b[e] = row[e];
  return b;
}

// ---------------- weight conversion kernels ----------------
__global__ void k_f32_to_f16(const float* __restrict__ src, _Float16* __restrict__ dst, int n) {
  int i = blockIdx.x * 256 + threadIdx.x;
  if (i < n) dst[i] = (_Float16)src[i];
}

__global__ void k_pad_f16(const float* __restrict__ src, _Float16* __restrict__ dst,
                          int rows, int ic, int oc) {
  int i = blockIdx.x * 256 + threadIdx.x;
  int n = rows * oc;
  if (i < n) {
    int r = i / oc, c = i % oc;
    dst[i] = (c < ic) ? (_Float16)src[r * ic + c] : (_Float16)0;
  }
}

// pcaps_w [co][ci][9][9] f32 -> Wp [p][co][ci] f16  (p = ky*9+kx)
__global__ void k_pcaps_tr(const float* __restrict__ src, _Float16* __restrict__ dst) {
  int i = blockIdx.x * 256 + threadIdx.x;
  if (i < 81 * 256 * 256) {
    int p = i / (256 * 256);
    int rem = i % (256 * 256);
    int co = rem / 256, ci = rem % 256;
    dst[i] = (_Float16)src[((size_t)co * 256 + ci) * 81 + p];
  }
}

// ---------------- conv1: 1->256 k9 s1 + ReLU -> h16 NHWC f16 ----------------
// M = 256*400 pixels, N = 256, K = 81 padded to 96. Block tile 128M x 128N.
__global__ void __launch_bounds__(256) k_conv1(const float* __restrict__ x,
                                               const _Float16* __restrict__ w1h, // [256][96]
                                               const float* __restrict__ bias,
                                               _Float16* __restrict__ h16) {
  __shared__ __align__(16) _Float16 Ash[128 * 96];
  __shared__ __align__(16) _Float16 Bsh[128 * 96];
  const int t = threadIdx.x;
  const int M0 = blockIdx.x * 128;
  const int co0 = blockIdx.y * 128;
  {
    const uint4* src = (const uint4*)(w1h + (size_t)co0 * 96);
    uint4* dst = (uint4*)Bsh;
#pragma unroll
    for (int i = 0; i < 6; ++i) dst[t + 256 * i] = src[t + 256 * i];
  }
  for (int j = 0; j < 48; ++j) {
    int idx = t + 256 * j;
    int m_l = idx / 96, k = idx % 96;
    int pix = M0 + m_l;
    int b = pix / 400, r2 = pix % 400;
    int oy = r2 / 20, ox = r2 % 20;
    float val = 0.f;
    if (k < 81) {
      int ky = k / 9, kx = k % 9;
      val = x[(size_t)b * 784 + (oy + ky) * 28 + (ox + kx)];
    }
    Ash[idx] = (_Float16)val;
  }
  __syncthreads();
  const int wave = t >> 5, lane = t & 31, nn = lane & 15, half = lane >> 4;
  v16h aF[3];
#pragma unroll
  for (int kc = 0; kc < 3; ++kc) aF[kc] = load_a_frag(Ash + wave * 16 * 96 + kc * 32, 96);
#pragma unroll
  for (int nt = 0; nt < 8; ++nt) {
    v8f acc;
#pragma unroll
    for (int i = 0; i < 8; ++i) acc[i] = 0.f;
#pragma unroll
    for (int kc = 0; kc < 3; ++kc) {
      v16h bF = load_b_frag(Bsh + (nt * 16) * 96 + kc * 32, 96);
      acc = wmma_f16(aF[kc], bF, acc);
    }
    int co = co0 + nt * 16 + nn;
    float bv = bias[co];
#pragma unroll
    for (int vv = 0; vv < 8; ++vv) {
      int pix = M0 + wave * 16 + 8 * half + vv;
      float v = fmaxf(acc[vv] + bv, 0.f);
      h16[(size_t)pix * 256 + co] = (_Float16)v;
    }
  }
}

// ---------------- primary caps conv: 256->256 k9 s2 + bias -> p_buf [b][co][36] f32 ----
// M = 256*36 = 9216, N = 256, K = 81*256. Block tile 64M x 256N, K-chunk = 64 (one p, 64 ci).
__global__ void __launch_bounds__(256) k_pcaps(const _Float16* __restrict__ h16,
                                               const _Float16* __restrict__ Wp, // [81][co][ci]
                                               const float* __restrict__ bias,
                                               float* __restrict__ p_buf) {
  __shared__ __align__(16) _Float16 Ash[64 * 64];
  __shared__ __align__(16) _Float16 Bsh[256 * 64];
  const int t = threadIdx.x;
  const int P0 = blockIdx.x * 64;
  const int wave = t >> 5, lane = t & 31, nn = lane & 15, half = lane >> 4;
  const int mrow = (wave & 3) * 16;
  const int cobase = (wave >> 2) * 128;
  v8f acc[8];
#pragma unroll
  for (int nt = 0; nt < 8; ++nt)
#pragma unroll
    for (int i = 0; i < 8; ++i) acc[nt][i] = 0.f;

  for (int p = 0; p < 81; ++p) {
    for (int cc = 0; cc < 4; ++cc) {
      int ci0 = cc * 64;
      __syncthreads();
      {
        const _Float16* wsrc = Wp + (size_t)p * 256 * 256 + ci0;
#pragma unroll
        for (int i = 0; i < 8; ++i) {
          int li = t + 256 * i;
          int co = li >> 3, q = li & 7;
          ((uint4*)Bsh)[li] = ((const uint4*)(wsrc + (size_t)co * 256))[q];
        }
        // hint next weight chunk toward L2
        __builtin_prefetch(wsrc + 256 * 256 + (size_t)t * 64, 0, 1);
      }
#pragma unroll
      for (int i = 0; i < 2; ++i) {
        int li = t + 256 * i;
        int m_l = li >> 3, q = li & 7;
        int pix = P0 + m_l;
        int b = pix / 36, s = pix % 36;
        int oy = s / 6, ox = s % 6;
        int iy = 2 * oy + p / 9, ix = 2 * ox + p % 9;
        ((uint4*)Ash)[li] =
            ((const uint4*)(h16 + ((size_t)b * 400 + iy * 20 + ix) * 256 + ci0))[q];
      }
      __syncthreads();
      v16h a0 = load_a_frag(Ash + mrow * 64, 64);
      v16h a1 = load_a_frag(Ash + mrow * 64 + 32, 64);
#pragma unroll
      for (int nt = 0; nt < 8; ++nt) {
        v16h b0 = load_b_frag(Bsh + (cobase + nt * 16) * 64, 64);
        acc[nt] = wmma_f16(a0, b0, acc[nt]);
        v16h b1 = load_b_frag(Bsh + (cobase + nt * 16) * 64 + 32, 64);
        acc[nt] = wmma_f16(a1, b1, acc[nt]);
      }
    }
  }
#pragma unroll
  for (int nt = 0; nt < 8; ++nt) {
    int co = cobase + nt * 16 + nn;
    float bv = bias[co];
#pragma unroll
    for (int vv = 0; vv < 8; ++vv) {
      int pix = P0 + mrow + 8 * half + vv;
      int b = pix / 36, s = pix % 36;
      p_buf[((size_t)b * 256 + co) * 36 + s] = acc[nt][vv] + bv;
    }
  }
}

// ---------------- squash groups of 8 -> u16 f16 ----------------
__global__ void __launch_bounds__(256) k_squash(const float* __restrict__ p_buf,
                                                _Float16* __restrict__ u16, int ngroups) {
  int g = blockIdx.x * 256 + threadIdx.x;
  if (g >= ngroups) return;
  const float4* src = (const float4*)(p_buf + (size_t)g * 8);
  float4 lo = src[0], hi = src[1];
  float v[8] = {lo.x, lo.y, lo.z, lo.w, hi.x, hi.y, hi.z, hi.w};
  float ns = 0.f;
#pragma unroll
  for (int i = 0; i < 8; ++i) ns += v[i] * v[i];
  float scale = (ns / (1.f + ns)) * rsqrtf(ns + 1e-16f);
  __align__(16) _Float16 o[8];
#pragma unroll
  for (int i = 0; i < 8; ++i) o[i] = (_Float16)(v[i] * scale);
  *((uint4*)(u16 + (size_t)g * 8)) = *((uint4*)o);
}

// ---------------- dynamic routing, one kernel per iteration ----------------
// Block = 16 batch rows x all 1152 routes; 10 waves, wave w owns class j=w.
// u_hat recomputed via WMMA (K=8 zero-padded to 32); c_ij folded into A scale.
__global__ void __launch_bounds__(320) k_routing(const _Float16* __restrict__ u16,
                                                 const _Float16* __restrict__ Wh, // [r][j][o][i]
                                                 const float* __restrict__ caps_bias, // [10][16]
                                                 float* __restrict__ b_ij,            // [B][1152][10]
                                                 float* __restrict__ v_out,           // caps_out
                                                 float* __restrict__ masked_out,
                                                 _Float16* __restrict__ dec16, int iter) {
  __shared__ __align__(16) _Float16 u_sh[16 * 8 * 8];        // [b][r][i]
  __shared__ __align__(16) _Float16 W_sh[8 * 10 * 16 * 8];   // [r][j][o][i]
  __shared__ float c_sh[16 * 8 * 10];
  __shared__ float v_sh[16 * 160];
  __shared__ float ns_sh[16 * 10];
  __shared__ int   arg_sh[16];
  const int t = threadIdx.x;
  const int wave = t >> 5, lane = t & 31, nn = lane & 15, half = lane >> 4;
  const int b0 = blockIdx.x * 16;
  const bool last = (iter == 2);

  v8f s_acc;
#pragma unroll
  for (int i = 0; i < 8; ++i) s_acc[i] = 0.f;

  for (int chunk = 0; chunk < 144; ++chunk) {
    int r0 = chunk * 8;
    __syncthreads();
    if (t < 128) {
      int b_l = t >> 3, r_l = t & 7;
      ((uint4*)u_sh)[t] =
          *((const uint4*)(u16 + (((size_t)(b0 + b_l)) * 1152 + r0 + r_l) * 8));
    }
#pragma unroll
    for (int i = 0; i < 4; ++i) {
      int li = t + 320 * i;
      int r_l = li / 160, q = li % 160;
      ((uint4*)W_sh)[li] = ((const uint4*)(Wh + ((size_t)(r0 + r_l)) * 1280))[q];
    }
    if (t < 128) {
      int b_l = t >> 3, r_l = t & 7;
      float* cd = c_sh + (b_l * 8 + r_l) * 10;
      if (iter == 0) {
#pragma unroll
        for (int j = 0; j < 10; ++j) cd[j] = 0.1f;
      } else {
        const float* bp = b_ij + (((size_t)(b0 + b_l)) * 1152 + r0 + r_l) * 10;
        float mx = bp[0];
#pragma unroll
        for (int j = 1; j < 10; ++j) mx = fmaxf(mx, bp[j]);
        float e[10], s = 0.f;
#pragma unroll
        for (int j = 0; j < 10; ++j) { e[j] = __expf(bp[j] - mx); s += e[j]; }
        float inv = 1.f / s;
#pragma unroll
        for (int j = 0; j < 10; ++j) cd[j] = e[j] * inv;
      }
    }
    __syncthreads();
    for (int r_l = 0; r_l < 8; ++r_l) {
      v16h a, bf;
#pragma unroll
      for (int e = 0; e < 16; ++e) { a[e] = (_Float16)0; bf[e] = (_Float16)0; }
      if (lane < 16) {
        float c = c_sh[(nn * 8 + r_l) * 10 + wave];
        const _Float16* up = u_sh + (nn * 8 + r_l) * 8;
        const _Float16* wp = W_sh + ((r_l * 10 + wave) * 16 + nn) * 8;
#pragma unroll
        for (int e = 0; e < 8; ++e) {
          a[e]  = (_Float16)(c * (float)up[e]);
          bf[e] = wp[e];
        }
      }
      s_acc = wmma_f16(a, bf, s_acc);
    }
  }

  // epilogue: s_j + bias -> squash over o (cross-lane over the 16 n-lanes)
  {
    float bsv = caps_bias[wave * 16 + nn];
    float sv[8], sq[8];
#pragma unroll
    for (int vv = 0; vv < 8; ++vv) { sv[vv] = s_acc[vv] + bsv; sq[vv] = sv[vv] * sv[vv]; }
#pragma unroll
    for (int m = 1; m <= 8; m <<= 1)
#pragma unroll
      for (int vv = 0; vv < 8; ++vv) sq[vv] += __shfl_xor(sq[vv], m, 32);
#pragma unroll
    for (int vv = 0; vv < 8; ++vv) {
      float ns = sq[vv];
      float scale = (ns / (1.f + ns)) * rsqrtf(ns + 1e-16f);
      float v = sv[vv] * scale;
      int row = 8 * half + vv;
      v_sh[row * 160 + wave * 16 + nn] = v;
      if (nn == 0) ns_sh[row * 10 + wave] = ns;
    }
  }
  __syncthreads();

  if (last) {
    for (int i = t; i < 2560; i += 320) {
      int b_l = i / 160, jo = i % 160;
      v_out[((size_t)(b0 + b_l)) * 160 + jo] = v_sh[b_l * 160 + jo];
    }
    if (t < 16) {
      int jm = 0;
      float best = ns_sh[t * 10];
#pragma unroll
      for (int j = 1; j < 10; ++j)
        if (ns_sh[t * 10 + j] > best) { best = ns_sh[t * 10 + j]; jm = j; }
      arg_sh[t] = jm;
#pragma unroll
      for (int j = 0; j < 10; ++j)
        masked_out[((size_t)(b0 + t)) * 10 + j] = (j == jm) ? 1.f : 0.f;
    }
    __syncthreads();
    for (int i = t; i < 2560; i += 320) {
      int b_l = i / 160, jo = i % 160, j = jo >> 4;
      float v = (j == arg_sh[b_l]) ? v_sh[i] : 0.f;
      dec16[((size_t)(b0 + b_l)) * 160 + jo] = (_Float16)v;
    }
  } else {
    // phase 2: b_ij += sum_o u_hat * v   (u_hat recomputed, unscaled)
    for (int chunk = 0; chunk < 144; ++chunk) {
      int r0 = chunk * 8;
      __syncthreads();
      if (t < 128) {
        int b_l = t >> 3, r_l = t & 7;
        ((uint4*)u_sh)[t] =
            *((const uint4*)(u16 + (((size_t)(b0 + b_l)) * 1152 + r0 + r_l) * 8));
      }
#pragma unroll
      for (int i = 0; i < 4; ++i) {
        int li = t + 320 * i;
        int r_l = li / 160, q = li % 160;
        ((uint4*)W_sh)[li] = ((const uint4*)(Wh + ((size_t)(r0 + r_l)) * 1280))[q];
      }
      __syncthreads();
      for (int r_l = 0; r_l < 8; ++r_l) {
        v16h a, bf;
#pragma unroll
        for (int e = 0; e < 16; ++e) { a[e] = (_Float16)0; bf[e] = (_Float16)0; }
        if (lane < 16) {
          const _Float16* up = u_sh + (nn * 8 + r_l) * 8;
          const _Float16* wp = W_sh + ((r_l * 10 + wave) * 16 + nn) * 8;
#pragma unroll
          for (int e = 0; e < 8; ++e) { a[e] = up[e]; bf[e] = wp[e]; }
        }
        v8f z;
#pragma unroll
        for (int i = 0; i < 8; ++i) z[i] = 0.f;
        v8f d = wmma_f16(a, bf, z);
        float tv[8];
#pragma unroll
        for (int vv = 0; vv < 8; ++vv) {
          int row = 8 * half + vv;
          tv[vv] = d[vv] * v_sh[row * 160 + wave * 16 + nn];
        }
#pragma unroll
        for (int m = 1; m <= 8; m <<= 1)
#pragma unroll
          for (int vv = 0; vv < 8; ++vv) tv[vv] += __shfl_xor(tv[vv], m, 32);
        if (nn == 0) {
#pragma unroll
          for (int vv = 0; vv < 8; ++vv) {
            int row = 8 * half + vv;
            size_t idx = (((size_t)(b0 + row)) * 1152 + r0 + r_l) * 10 + wave;
            if (iter == 0) b_ij[idx] = tv[vv];
            else           b_ij[idx] += tv[vv];
          }
        }
      }
    }
  }
}

// ---------------- generic WMMA GEMM for the FC decoder ----------------
// out = act(A[M=256,K] @ W[N,K]^T + bias); act 0=relu, 1=sigmoid.
__global__ void __launch_bounds__(128) k_fc(const _Float16* __restrict__ A,
                                            const _Float16* __restrict__ W,
                                            const float* __restrict__ bias,
                                            _Float16* __restrict__ out16,
                                            float* __restrict__ out32,
                                            int N, int K, int act) {
  const int wave = threadIdx.x >> 5, lane = threadIdx.x & 31;
  const int nn = lane & 15, half = lane >> 4;
  const int id = blockIdx.x * 4 + wave;
  const int m0 = (id & 15) * 16;
  const int n0 = (id >> 4) * 16;
  v8f acc;
#pragma unroll
  for (int i = 0; i < 8; ++i) acc[i] = 0.f;
  for (int k0 = 0; k0 < K; k0 += 32) {
    const _Float16* arow = A + (size_t)(m0 + nn) * K + k0;
    v16h a;
#pragma unroll
    for (int ep = 0; ep < 8; ++ep) {
      int e = ep * 2;
      int k = half * 16 + e;
      a[e] = arow[k];
      a[e + 1] = arow[k + 1];
    }
    const _Float16* brow = W + (size_t)(n0 + nn) * K + k0 + half * 16;
    v16h b;
#pragma unroll
    for (int e = 0; e < 16; ++e) b[e] = brow[e];
    acc = wmma_f16(a, b, acc);
  }
  float bs = bias[n0 + nn];
#pragma unroll
  for (int vv = 0; vv < 8; ++vv) {
    int row = m0 + 8 * half + vv;
    float x = acc[vv] + bs;
    x = act ? (1.f / (1.f + __expf(-x))) : fmaxf(x, 0.f);
    if (out16) out16[(size_t)row * N + n0 + nn] = (_Float16)x;
    if (out32) out32[(size_t)row * N + n0 + nn] = x;
  }
}

extern "C" void kernel_launch(void* const* d_in, const int* in_sizes, int n_in,
                              void* d_out, int out_size, void* d_ws, size_t ws_size,
                              hipStream_t stream) {
  (void)in_sizes; (void)n_in; (void)out_size; (void)ws_size;
  const float* x         = (const float*)d_in[0];
  const float* conv1_w   = (const float*)d_in[1];
  const float* conv1_b   = (const float*)d_in[2];
  const float* pcaps_w   = (const float*)d_in[3];
  const float* pcaps_b   = (const float*)d_in[4];
  const float* W_caps    = (const float*)d_in[5];
  const float* caps_bias = (const float*)d_in[6];
  const float* fc1_w = (const float*)d_in[7];
  const float* fc1_b = (const float*)d_in[8];
  const float* fc2_w = (const float*)d_in[9];
  const float* fc2_b = (const float*)d_in[10];
  const float* fc3_w = (const float*)d_in[11];
  const float* fc3_b = (const float*)d_in[12];
  float* out = (float*)d_out;  // [caps 40960 | rec 200704 | masked 2560]

  char* base = (char*)d_ws;
  size_t off = 0;
  auto alloc = [&](size_t bytes) -> void* {
    void* p = base + off;
    off = (off + bytes + 255) & ~(size_t)255;
    return p;
  };
  _Float16* h16   = (_Float16*)alloc((size_t)256 * 400 * 256 * 2); // conv1 out NHWC f16
  _Float16* Wp    = (_Float16*)alloc((size_t)81 * 256 * 256 * 2);  // pcaps weights [p][co][ci]
  _Float16* w1h   = (_Float16*)alloc((size_t)256 * 96 * 2);        // conv1 w padded
  float*    pbuf  = (float*)   alloc((size_t)256 * 256 * 36 * 4);  // pcaps out f32
  _Float16* u16   = (_Float16*)alloc((size_t)256 * 1152 * 8 * 2);  // squashed u f16
  _Float16* Wh    = (_Float16*)alloc((size_t)1152 * 1280 * 2);     // W_caps f16
  float*    bij   = (float*)   alloc((size_t)256 * 1152 * 10 * 4); // routing logits
  _Float16* dec16 = (_Float16*)alloc((size_t)256 * 160 * 2);
  _Float16* a1    = (_Float16*)alloc((size_t)256 * 512 * 2);
  _Float16* a2    = (_Float16*)alloc((size_t)256 * 1024 * 2);
  _Float16* fc1h  = (_Float16*)alloc((size_t)512 * 160 * 2);
  _Float16* fc2h  = (_Float16*)alloc((size_t)1024 * 512 * 2);
  _Float16* fc3h  = (_Float16*)alloc((size_t)784 * 1024 * 2);

  // weight conversions
  k_pad_f16 <<<96, 256, 0, stream>>>(conv1_w, w1h, 256, 81, 96);
  k_pcaps_tr<<<20736, 256, 0, stream>>>(pcaps_w, Wp);
  k_f32_to_f16<<<(1474560 + 255) / 256, 256, 0, stream>>>(W_caps, Wh, 1474560);
  k_f32_to_f16<<<(81920 + 255) / 256, 256, 0, stream>>>(fc1_w, fc1h, 81920);
  k_f32_to_f16<<<(524288 + 255) / 256, 256, 0, stream>>>(fc2_w, fc2h, 524288);
  k_f32_to_f16<<<(802816 + 255) / 256, 256, 0, stream>>>(fc3_w, fc3h, 802816);

  // main pipeline
  k_conv1<<<dim3(800, 2), 256, 0, stream>>>(x, w1h, conv1_b, h16);
  k_pcaps<<<144, 256, 0, stream>>>(h16, Wp, pcaps_b, pbuf);
  k_squash<<<1152, 256, 0, stream>>>(pbuf, u16, 294912);
  for (int it = 0; it < 3; ++it)
    k_routing<<<16, 320, 0, stream>>>(u16, Wh, caps_bias, bij,
                                      out, out + 241664, dec16, it);
  // reconstruction decoder
  k_fc<<<128, 128, 0, stream>>>(dec16, fc1h, fc1_b, a1, (float*)nullptr, 512, 160, 0);
  k_fc<<<256, 128, 0, stream>>>(a1, fc2h, fc2_b, a2, (float*)nullptr, 1024, 512, 0);
  k_fc<<<196, 128, 0, stream>>>(a2, fc3h, fc3_b, (_Float16*)nullptr, out + 40960, 784, 1024, 1);
}
